// PostEncode_43224550867683
// MI455X (gfx1250) — matrix operand order repaired and checked by reading, not compiled, not run on hardware
//
#include <hip/hip_runtime.h>
#include <hip/hip_bf16.h>
#include <math.h>

#define NNODES 16384
#define LHIST  50
#define LPAD   64
#define EDIM   64
#define CEDIM  128

typedef __bf16 bf16_t;
typedef bf16_t bf16x16 __attribute__((ext_vector_type(16)));
typedef float  floatx8 __attribute__((ext_vector_type(8)));

union FragAB { bf16x16 v; uint4 q[2]; };
union FragC  { floatx8 v; float f[8]; };

__device__ inline floatx8 wmma_bf16(const FragAB& a, const FragAB& b, floatx8 c) {
    return __builtin_amdgcn_wmma_f32_16x16x32_bf16(false, a.v, false, b.v, (short)0, c, false, false);
}

// A fragment: 16x32 bf16 tile from row-major buffer, row stride ld (elements).
// lane m = lane&15 selects row, half = lane>>4 selects K sub-block per ISA layout.
__device__ inline FragAB load_frag_a(const bf16_t* buf, int ld, int row0, int kbase, int lane) {
    int m = lane & 15, half = lane >> 4;
    const bf16_t* r = buf + (row0 + m) * ld + kbase + half * 8;
    FragAB f;
    f.q[0] = *(const uint4*)(r);
    f.q[1] = *(const uint4*)(r + 16);
    return f;
}

// B fragment: 32x16 bf16 from PRE-TRANSPOSED weights Wt[nout][K], row stride ld.
// lane n = lane&15 selects output column, half selects K sub-block per ISA layout (same as A).
__device__ inline FragAB load_frag_b(const bf16_t* wt, int ld, int n0, int kbase, int lane) {
    int n = lane & 15, half = lane >> 4;
    const bf16_t* r = wt + (n0 + n) * ld + kbase + half * 8;
    FragAB f;
    f.q[0] = *(const uint4*)(r);
    f.q[1] = *(const uint4*)(r + 16);
    return f;
}

// ---------------- prep: transpose f32 [K][Nc] -> bf16 [Nc][K] ----------------
__global__ void k_prep(const float* __restrict__ src, bf16_t* __restrict__ dst, int K, int Nc) {
    int i = blockIdx.x * blockDim.x + threadIdx.x;
    if (i < K * Nc) {
        int k = i / Nc, n = i % Nc;
        dst[n * K + k] = (bf16_t)src[i];
    }
}

// ---------------- post = relu(content_emb[prc] @ We + b) ----------------
__global__ __launch_bounds__(32) void k_post(const float* __restrict__ cemb,
                                             const int* __restrict__ prc,
                                             const bf16_t* __restrict__ Wet,
                                             const float* __restrict__ Web,
                                             float* __restrict__ postF) {
    __shared__ bf16_t arow[16][CEDIM];
    int lane = threadIdx.x;
    int m = lane & 15, half = lane >> 4;
    int row0 = blockIdx.x * 16;

    const float* src = cemb + (long)prc[row0 + m] * CEDIM + half * 64;
    #pragma unroll
    for (int c = 0; c < 64; c += 4) {
        float4 v = *(const float4*)(src + c);
        bf16_t* d = &arow[m][half * 64 + c];
        d[0] = (bf16_t)v.x; d[1] = (bf16_t)v.y; d[2] = (bf16_t)v.z; d[3] = (bf16_t)v.w;
    }
    __syncthreads();

    FragC acc[4];
    #pragma unroll
    for (int nt = 0; nt < 4; nt++)
        #pragma unroll
        for (int j = 0; j < 8; j++) acc[nt].f[j] = 0.f;

    #pragma unroll
    for (int kt = 0; kt < 4; kt++) {
        FragAB a = load_frag_a(&arow[0][0], CEDIM, 0, kt * 32, lane);
        #pragma unroll
        for (int nt = 0; nt < 4; nt++) {
            FragAB b = load_frag_b(Wet, CEDIM, nt * 16, kt * 32, lane);
            acc[nt].v = wmma_bf16(a, b, acc[nt].v);
        }
    }

    int n = lane & 15;
    #pragma unroll
    for (int nt = 0; nt < 4; nt++) {
        int col = nt * 16 + n;
        float bias = Web[col];
        #pragma unroll
        for (int j = 0; j < 8; j++) {
            int row = row0 + j + 8 * half;
            float v = acc[nt].f[j] + bias;
            postF[row * EDIM + col] = v > 0.f ? v : 0.f;
        }
    }
}

// ---------------- fused per-node neighbor pipeline ----------------
// Wall = contiguous [W1t(8192) | W2t(4096) | A1t(8192) | A2t(4096)] bf16
__global__ __launch_bounds__(128) void k_node(
    const float* __restrict__ u2e, const float* __restrict__ r2e,
    const int* __restrict__ puh, const int* __restrict__ prh,
    const int* __restrict__ lens,
    const bf16_t* __restrict__ Wall,
    const float* __restrict__ W1b, const float* __restrict__ W2b,
    const float* __restrict__ A1b, const float* __restrict__ A2b,
    const float* __restrict__ A3w, const float* __restrict__ A3b,
    const float* __restrict__ postF,
    float* __restrict__ histF)
{
    __shared__ bf16_t xin[LPAD][CEDIM];   // concat [u|r] ; later reused for a2
    __shared__ bf16_t xb [LPAD][EDIM];    // x ; later reused for a1
    __shared__ bf16_t ob [LPAD][EDIM];    // o
    __shared__ bf16_t wlds[24576];        // W1t | W2t | A1t | A2t (48 KB)
    __shared__ bf16_t postL[EDIM];
    __shared__ float  a3L[EDIM];
    __shared__ float  logitL[LPAD];
    __shared__ float  attL[LPAD];
    __shared__ float  histL[EDIM];
    __shared__ int    puL[LPAD], prL[LPAD];
    __shared__ float  red[2];

    const bf16_t* W1l = wlds;
    const bf16_t* W2l = wlds + 8192;
    const bf16_t* A1l = wlds + 12288;
    const bf16_t* A2l = wlds + 20480;

    int node = blockIdx.x;
    int t = threadIdx.x;
    int lane = t & 31;
    int wv = t >> 5;
    int m0 = wv * 16;
    int n = lane & 15, half = lane >> 4;
    int len = lens[node];

    if (t < LHIST) { puL[t] = puh[node * LHIST + t]; prL[t] = prh[node * LHIST + t]; }
    if (t < EDIM)  { postL[t] = (bf16_t)postF[node * EDIM + t]; a3L[t] = A3w[t]; histL[t] = 0.f; }

    // stage all weights into LDS (3072 uint4 / 128 threads = 24 each)
    {
        const uint4* ws = (const uint4*)Wall;
        uint4* wd = (uint4*)wlds;
        #pragma unroll
        for (int i = 0; i < 24; i++) wd[t + i * 128] = ws[t + i * 128];
    }
    __syncthreads();

    // gather neighbor embeddings -> bf16 LDS, zero the padded rows (l >= 50)
    for (int i = t; i < LPAD * (CEDIM / 4); i += 128) {
        int l = i >> 5;              // 32 float4-groups per row
        int c4 = (i & 31) * 4;
        float4 v = make_float4(0.f, 0.f, 0.f, 0.f);
        if (l < LHIST) {
            if (c4 < 64) v = *(const float4*)(u2e + (long)puL[l] * EDIM + c4);
            else         v = *(const float4*)(r2e + (long)prL[l] * EDIM + (c4 - 64));
        }
        bf16_t* d = &xin[l][c4];
        d[0] = (bf16_t)v.x; d[1] = (bf16_t)v.y; d[2] = (bf16_t)v.z; d[3] = (bf16_t)v.w;
    }
    __syncthreads();

    // ---- x = relu([u|r] @ W1 + b1) : K=128
    FragC acc[4];
    #pragma unroll
    for (int nt = 0; nt < 4; nt++)
        #pragma unroll
        for (int j = 0; j < 8; j++) acc[nt].f[j] = 0.f;
    #pragma unroll
    for (int kt = 0; kt < 4; kt++) {
        FragAB a = load_frag_a(&xin[0][0], CEDIM, m0, kt * 32, lane);
        #pragma unroll
        for (int nt = 0; nt < 4; nt++) {
            FragAB b = load_frag_b(W1l, CEDIM, nt * 16, kt * 32, lane);
            acc[nt].v = wmma_bf16(a, b, acc[nt].v);
        }
    }
    #pragma unroll
    for (int nt = 0; nt < 4; nt++) {
        float bias = W1b[nt * 16 + n];
        #pragma unroll
        for (int j = 0; j < 8; j++) {
            float v = acc[nt].f[j] + bias; v = v > 0.f ? v : 0.f;
            xb[m0 + j + 8 * half][nt * 16 + n] = (bf16_t)v;
        }
    }
    __syncthreads();

    // ---- o = relu(x @ W2 + b2) : K=64 ; keep o in registers for weighted sum
    FragC oacc[4];
    #pragma unroll
    for (int nt = 0; nt < 4; nt++)
        #pragma unroll
        for (int j = 0; j < 8; j++) oacc[nt].f[j] = 0.f;
    #pragma unroll
    for (int kt = 0; kt < 2; kt++) {
        FragAB a = load_frag_a(&xb[0][0], EDIM, m0, kt * 32, lane);
        #pragma unroll
        for (int nt = 0; nt < 4; nt++) {
            FragAB b = load_frag_b(W2l, EDIM, nt * 16, kt * 32, lane);
            oacc[nt].v = wmma_bf16(a, b, oacc[nt].v);
        }
    }
    #pragma unroll
    for (int nt = 0; nt < 4; nt++) {
        float bias = W2b[nt * 16 + n];
        #pragma unroll
        for (int j = 0; j < 8; j++) {
            float v = oacc[nt].f[j] + bias; v = v > 0.f ? v : 0.f;
            oacc[nt].f[j] = v;                                  // keep relu'd o
            ob[m0 + j + 8 * half][nt * 16 + n] = (bf16_t)v;
        }
    }
    __syncthreads();

    // ---- a = relu([o|post] @ A1 + bA1) : K=128 (k>=64 is broadcast post row)
    #pragma unroll
    for (int nt = 0; nt < 4; nt++)
        #pragma unroll
        for (int j = 0; j < 8; j++) acc[nt].f[j] = 0.f;
    #pragma unroll
    for (int kt = 0; kt < 4; kt++) {
        FragAB a;
        if (kt < 2) {
            a = load_frag_a(&ob[0][0], EDIM, m0, kt * 32, lane);
        } else {
            const bf16_t* r = postL + (kt - 2) * 32 + half * 8;  // same for all rows
            a.q[0] = *(const uint4*)(r);
            a.q[1] = *(const uint4*)(r + 16);
        }
        #pragma unroll
        for (int nt = 0; nt < 4; nt++) {
            FragAB b = load_frag_b(A1l, CEDIM, nt * 16, kt * 32, lane);
            acc[nt].v = wmma_bf16(a, b, acc[nt].v);
        }
    }
    #pragma unroll
    for (int nt = 0; nt < 4; nt++) {
        float bias = A1b[nt * 16 + n];
        #pragma unroll
        for (int j = 0; j < 8; j++) {
            float v = acc[nt].f[j] + bias; v = v > 0.f ? v : 0.f;
            xb[m0 + j + 8 * half][nt * 16 + n] = (bf16_t)v;      // a1 -> xb
        }
    }
    __syncthreads();

    // ---- a2 = relu(a1 @ A2 + bA2) : K=64 ; a2 -> xin (reuse)
    #pragma unroll
    for (int nt = 0; nt < 4; nt++)
        #pragma unroll
        for (int j = 0; j < 8; j++) acc[nt].f[j] = 0.f;
    #pragma unroll
    for (int kt = 0; kt < 2; kt++) {
        FragAB a = load_frag_a(&xb[0][0], EDIM, m0, kt * 32, lane);
        #pragma unroll
        for (int nt = 0; nt < 4; nt++) {
            FragAB b = load_frag_b(A2l, EDIM, nt * 16, kt * 32, lane);
            acc[nt].v = wmma_bf16(a, b, acc[nt].v);
        }
    }
    #pragma unroll
    for (int nt = 0; nt < 4; nt++) {
        float bias = A2b[nt * 16 + n];
        #pragma unroll
        for (int j = 0; j < 8; j++) {
            float v = acc[nt].f[j] + bias; v = v > 0.f ? v : 0.f;
            xin[m0 + j + 8 * half][nt * 16 + n] = (bf16_t)v;
        }
    }
    __syncthreads();

    // ---- logits = a2 @ A3 + bA3 , masked by length
    if (t < LPAD) {
        float s = A3b[0];
        #pragma unroll 8
        for (int e = 0; e < EDIM; e++) s += (float)xin[t][e] * a3L[e];
        logitL[t] = (t < len) ? s : -INFINITY;
    }
    __syncthreads();

    // ---- softmax over valid slots
    if (t == 0) {
        float mx = -INFINITY;
        for (int l = 0; l < LPAD; l++) mx = fmaxf(mx, logitL[l]);
        red[0] = mx;
    }
    __syncthreads();
    if (t < LPAD) attL[t] = (t < len) ? __expf(logitL[t] - red[0]) : 0.f;
    __syncthreads();
    if (t == 0) {
        float s = 0.f;
        for (int l = 0; l < LPAD; l++) s += attL[l];
        red[1] = s;
    }
    __syncthreads();
    if (t < LPAD) attL[t] = attL[t] / red[1];
    __syncthreads();

    // ---- hist = sum_l att[l] * o[l]  (o in registers, reduce via LDS atomics)
    #pragma unroll
    for (int nt = 0; nt < 4; nt++) {
        float p = 0.f;
        #pragma unroll
        for (int j = 0; j < 8; j++) p += attL[m0 + j + 8 * half] * oacc[nt].f[j];
        atomicAdd(&histL[nt * 16 + n], p);
    }
    __syncthreads();
    if (t < EDIM) histF[node * EDIM + t] = histL[t];
}

// ---------------- out = relu([hist|post] @ Ow + bOw) ----------------
__global__ __launch_bounds__(32) void k_final(const float* __restrict__ histF,
                                              const float* __restrict__ postF,
                                              const bf16_t* __restrict__ Owt,
                                              const float* __restrict__ Owb,
                                              float* __restrict__ out) {
    __shared__ bf16_t arow[16][CEDIM];
    int lane = threadIdx.x;
    int m = lane & 15, half = lane >> 4;
    int row0 = blockIdx.x * 16;

    const float* src = (half == 0) ? (histF + (long)(row0 + m) * EDIM)
                                   : (postF + (long)(row0 + m) * EDIM);
    #pragma unroll
    for (int c = 0; c < 64; c += 4) {
        float4 v = *(const float4*)(src + c);
        bf16_t* d = &arow[m][half * 64 + c];
        d[0] = (bf16_t)v.x; d[1] = (bf16_t)v.y; d[2] = (bf16_t)v.z; d[3] = (bf16_t)v.w;
    }
    __syncthreads();

    FragC acc[4];
    #pragma unroll
    for (int nt = 0; nt < 4; nt++)
        #pragma unroll
        for (int j = 0; j < 8; j++) acc[nt].f[j] = 0.f;
    #pragma unroll
    for (int kt = 0; kt < 4; kt++) {
        FragAB a = load_frag_a(&arow[0][0], CEDIM, 0, kt * 32, lane);
        #pragma unroll
        for (int nt = 0; nt < 4; nt++) {
            FragAB b = load_frag_b(Owt, CEDIM, nt * 16, kt * 32, lane);
            acc[nt].v = wmma_bf16(a, b, acc[nt].v);
        }
    }
    int n = lane & 15;
    #pragma unroll
    for (int nt = 0; nt < 4; nt++) {
        int col = nt * 16 + n;
        float bias = Owb[col];
        #pragma unroll
        for (int j = 0; j < 8; j++) {
            int row = row0 + j + 8 * half;
            float v = acc[nt].f[j] + bias;
            out[row * EDIM + col] = v > 0.f ? v : 0.f;
        }
    }
}

extern "C" void kernel_launch(void* const* d_in, const int* in_sizes, int n_in,
                              void* d_out, int out_size, void* d_ws, size_t ws_size,
                              hipStream_t stream) {
    const float* u2e   = (const float*)d_in[0];
    const float* r2e   = (const float*)d_in[1];
    const float* cemb  = (const float*)d_in[2];
    const float* We_w  = (const float*)d_in[3];
    const float* We_b  = (const float*)d_in[4];
    const float* W1_w  = (const float*)d_in[5];
    const float* W1_b  = (const float*)d_in[6];
    const float* W2_w  = (const float*)d_in[7];
    const float* W2_b  = (const float*)d_in[8];
    const float* A1_w  = (const float*)d_in[9];
    const float* A1_b  = (const float*)d_in[10];
    const float* A2_w  = (const float*)d_in[11];
    const float* A2_b  = (const float*)d_in[12];
    const float* A3_w  = (const float*)d_in[13];
    const float* A3_b  = (const float*)d_in[14];
    const float* Ow_w  = (const float*)d_in[15];
    const float* Ow_b  = (const float*)d_in[16];
    const int*   puh   = (const int*)d_in[18];
    const int*   prh   = (const int*)d_in[19];
    const int*   lens  = (const int*)d_in[20];
    const int*   prc   = (const int*)d_in[21];

    char* ws = (char*)d_ws;
    size_t off = 0;
    bf16_t* Wet  = (bf16_t*)(ws + off); off += 128 * 64 * sizeof(bf16_t);
    bf16_t* Wall = (bf16_t*)(ws + off);                       // W1t|W2t|A1t|A2t contiguous
    bf16_t* W1t  = (bf16_t*)(ws + off); off += 128 * 64 * sizeof(bf16_t);
    bf16_t* W2t  = (bf16_t*)(ws + off); off +=  64 * 64 * sizeof(bf16_t);
    bf16_t* A1t  = (bf16_t*)(ws + off); off += 128 * 64 * sizeof(bf16_t);
    bf16_t* A2t  = (bf16_t*)(ws + off); off +=  64 * 64 * sizeof(bf16_t);
    bf16_t* Owt  = (bf16_t*)(ws + off); off += 128 * 64 * sizeof(bf16_t);
    off = (off + 255) & ~(size_t)255;
    float*  postF = (float*)(ws + off);  off += (size_t)NNODES * EDIM * sizeof(float);
    float*  histF = (float*)(ws + off);  off += (size_t)NNODES * EDIM * sizeof(float);

    // weight prep (transpose + f32->bf16)
    k_prep<<<(128 * 64 + 255) / 256, 256, 0, stream>>>(We_w, Wet, 128, 64);
    k_prep<<<(128 * 64 + 255) / 256, 256, 0, stream>>>(W1_w, W1t, 128, 64);
    k_prep<<<( 64 * 64 + 255) / 256, 256, 0, stream>>>(W2_w, W2t,  64, 64);
    k_prep<<<(128 * 64 + 255) / 256, 256, 0, stream>>>(A1_w, A1t, 128, 64);
    k_prep<<<( 64 * 64 + 255) / 256, 256, 0, stream>>>(A2_w, A2t,  64, 64);
    k_prep<<<(128 * 64 + 255) / 256, 256, 0, stream>>>(Ow_w, Owt, 128, 64);

    k_post<<<NNODES / 16, 32, 0, stream>>>(cemb, prc, Wet, We_b, postF);

    k_node<<<NNODES, 128, 0, stream>>>(u2e, r2e, puh, prh, lens,
                                       Wall, W1_b, W2_b, A1_b, A2_b,
                                       A3_w, A3_b, postF, histF);

    k_final<<<NNODES / 16, 32, 0, stream>>>(histF, postF, Owt, Ow_b, (float*)d_out);
}
